// MapNet_52046413693246
// MI455X (gfx1250) — matrix operand order repaired and checked by reading, not compiled, not run on hardware
//
#include <hip/hip_runtime.h>
#include <hip/hip_bf16.h>

typedef unsigned short ushort_t;
typedef __attribute__((ext_vector_type(8)))  float   v8f;
typedef __attribute__((ext_vector_type(16))) __bf16  v16bf;

union BF16x16 {
    v16bf v;
    uint4 q[2];
};

__device__ __forceinline__ ushort_t f2bf(float f) {
    unsigned u = __float_as_uint(f);
    unsigned r = u + 0x7FFFu + ((u >> 16) & 1u);   // round-to-nearest-even
    return (ushort_t)(r >> 16);
}
__device__ __forceinline__ float bf2f(ushort_t h) {
    return __uint_as_float(((unsigned)h) << 16);
}

// ---------------------------------------------------------------------------
// Convert fp32 [128,128] weight matrices into bf16 tiles pre-swizzled into the
// WMMA B-operand register layout this code uses (contiguous k, same as A):
//   tile t = nt*4 + kt  (nt: 16-col group, kt: 32-K group)
//   lane l, element e:  K = kt*32 + (l>>4)*16 + e,  N = nt*16 + (l&15)
// so the GEMM kernel can fetch B with two contiguous 16B LDS reads per lane.
// ---------------------------------------------------------------------------
__global__ __launch_bounds__(256) void prep_weights(const float* __restrict__ src,
                                                    ushort_t* __restrict__ dst,
                                                    int nmat) {
    long idx = (long)blockIdx.x * 256 + threadIdx.x;
    long tot = (long)nmat * 16384;
    if (idx >= tot) return;
    int m  = (int)(idx >> 14);
    int r  = (int)(idx & 16383);
    int e  = r & 15;
    int l  = (r >> 4) & 31;
    int t  = r >> 9;          // 0..31
    int kt = t & 3;
    int nt = t >> 2;
    int k  = kt * 32 + (l >> 4) * 16 + e;
    int n  = nt * 16 + (l & 15);
    dst[idx] = f2bf(src[(long)m * 16384 + k * 128 + n]);
}

// ---------------------------------------------------------------------------
// Core kernel: out[u[r]] (+)= bf16(feat[v[r]]) @ Wswz  for 16-row tiles.
// One wave -> 16 rows x 128 cols (8 f32 accum tiles, 32 x v_wmma_f32_16x16x32_bf16).
// Weight staged in LDS (32KB) once per block; B read via ds_load_b128.
// Per kt chunk, ALL 8 B tiles are loaded into live registers before the WMMA
// chain, forcing the scheduler into the batched issue + decrementing-wait
// pattern (each WMMA waits only for its own loads, rest remain in flight).
// A (global gather) stays double-buffered across kt.
// eu/ev == nullptr means identity (dense GEMM over rows).
// ---------------------------------------------------------------------------
template <bool ATOMIC>
__global__ __launch_bounds__(256) void wmma_edge_gemm(const ushort_t* __restrict__ feat,
                                                      const ushort_t* __restrict__ Wswz,
                                                      const int* __restrict__ eu,
                                                      const int* __restrict__ ev,
                                                      float* __restrict__ out,
                                                      int nrows) {
    __shared__ ushort_t wlds[128 * 128];           // 32 KB of the 320 KB/WGP
    {   // cooperative 32KB global->LDS copy (16B per transaction)
        const uint4* ws4 = (const uint4*)Wswz;
        uint4*       wd4 = (uint4*)wlds;
        for (int i = threadIdx.x; i < 2048; i += 256) wd4[i] = ws4[i];
    }
    __syncthreads();

    const int wave = threadIdx.x >> 5;
    const int lane = threadIdx.x & 31;
    const int half = lane >> 4;      // 0: lanes 0-15, 1: lanes 16-31
    const int l15  = lane & 15;
    const long base = ((long)blockIdx.x * 8 + wave) * 16;
    if (base >= nrows) return;

    // Gather source row for this lane's slice of the A matrix (ISA 16x32 layout)
    int arow = (int)base + l15;
    int asrc = (arow < nrows) ? (ev ? ev[arow] : arow) : 0;
    const ushort_t* aptr = feat + (long)asrc * 128;

    // Hoist the scatter/store row indices so these gathers overlap the WMMAs.
    int um[8];
#pragma unroll
    for (int g = 0; g < 8; ++g) {
        int r = (int)base + half * 8 + g;
        um[g] = (r < nrows) ? (eu ? eu[r] : r) : -1;
    }

    v8f zero;
#pragma unroll
    for (int e = 0; e < 8; ++e) zero[e] = 0.0f;
    v8f acc[8];
#pragma unroll
    for (int nt = 0; nt < 8; ++nt) acc[nt] = zero;

    // A 16x32 bf16 layout: lanes 0-15 hold K {0..15}, lanes 16-31 {16..31}
    const int kA = half * 16;
    BF16x16 A[2];
    A[0].q[0] = *(const uint4*)(aptr + kA);
    A[0].q[1] = *(const uint4*)(aptr + kA + 8);

    const ushort_t* blane = wlds + (lane << 4);

#pragma unroll
    for (int kt = 0; kt < 4; ++kt) {
        if (kt < 3) {   // prefetch next A chunk (global, from L2-resident feat)
            const int k0 = (kt + 1) * 32 + kA;
            A[(kt + 1) & 1].q[0] = *(const uint4*)(aptr + k0);
            A[(kt + 1) & 1].q[1] = *(const uint4*)(aptr + k0 + 8);
        }
        const BF16x16& Ac = A[kt & 1];

        // Batch-load all 8 B tiles of this kt: 16 ds_load_b128 in flight at once.
        BF16x16 Bt[8];
#pragma unroll
        for (int nt = 0; nt < 8; ++nt) {
            const ushort_t* bp = blane + ((((nt * 4) + kt) * 32) << 4);
            Bt[nt].q[0] = *(const uint4*)(bp);
            Bt[nt].q[1] = *(const uint4*)(bp + 8);
        }
#pragma unroll
        for (int nt = 0; nt < 8; ++nt) {
            acc[nt] = __builtin_amdgcn_wmma_f32_16x16x32_bf16(
                false, Ac.v, false, Bt[nt].v, (short)0, acc[nt], false, false);
        }
    }

    // Epilogue: C/D layout — VGPR g, lanes 0-15 -> M=g, lanes 16-31 -> M=g+8, N=l15
#pragma unroll
    for (int g = 0; g < 8; ++g) {
        if (um[g] >= 0) {
            float* op = out + (long)um[g] * 128 + l15;
#pragma unroll
            for (int nt = 0; nt < 8; ++nt) {
                if (ATOMIC) {
                    unsafeAtomicAdd(op + nt * 16, acc[nt][g]);   // global_atomic_add_f32
                } else {
                    op[nt * 16] = acc[nt][g];
                }
            }
        }
    }
}

// ---------------------------------------------------------------------------
// GroupNorm(1, C) + ReLU over each 128-ch row; one wave32 per row.
// ---------------------------------------------------------------------------
__global__ __launch_bounds__(256) void gn_relu(const float* __restrict__ temp,
                                               const float* __restrict__ gamma,
                                               const float* __restrict__ beta,
                                               ushort_t* __restrict__ outb,
                                               int nrows) {
    const int wave = threadIdx.x >> 5;
    const int lane = threadIdx.x & 31;
    const int row  = blockIdx.x * 8 + wave;
    if (row >= nrows) return;
    const float4 x = *(const float4*)(temp + (long)row * 128 + lane * 4);
    float s  = x.x + x.y + x.z + x.w;
    float ss = x.x * x.x + x.y * x.y + x.z * x.z + x.w * x.w;
#pragma unroll
    for (int off = 16; off >= 1; off >>= 1) {
        s  += __shfl_xor(s,  off, 32);
        ss += __shfl_xor(ss, off, 32);
    }
    const float mu  = s * (1.0f / 128.0f);
    const float var = ss * (1.0f / 128.0f) - mu * mu;
    const float rs  = rsqrtf(var + 1e-5f);
    const int c0 = lane * 4;
    float vals[4] = {x.x, x.y, x.z, x.w};
    ushort_t o[4];
#pragma unroll
    for (int j = 0; j < 4; ++j) {
        float v = (vals[j] - mu) * rs * gamma[c0 + j] + beta[c0 + j];
        o[j] = f2bf(fmaxf(v, 0.0f));
    }
    *(ushort2*)(outb + (long)row * 128 + c0)     = make_ushort2(o[0], o[1]);
    *(ushort2*)(outb + (long)row * 128 + c0 + 2) = make_ushort2(o[2], o[3]);
}

// ---------------------------------------------------------------------------
// Pointwise kernels
// ---------------------------------------------------------------------------
// First (K=2) MLP layers for both input branches -> bf16 hidden activations.
__global__ __launch_bounds__(256) void input_stage1(const float* __restrict__ feats,
                                                    const float* __restrict__ ctrs,
                                                    const float* __restrict__ inW1,
                                                    const float* __restrict__ inb1,
                                                    const float* __restrict__ segW1,
                                                    const float* __restrict__ segb1,
                                                    ushort_t* __restrict__ hin,
                                                    ushort_t* __restrict__ hseg,
                                                    int total) {
    long idx = (long)blockIdx.x * 256 + threadIdx.x;
    if (idx >= total) return;
    int  c    = (int)(idx & 127);
    long node = idx >> 7;
    float c0 = ctrs[node * 2], c1 = ctrs[node * 2 + 1];
    float f0 = feats[node * 2], f1 = feats[node * 2 + 1];
    hin[idx]  = f2bf(fmaxf(c0 * inW1[c]  + c1 * inW1[128 + c]  + inb1[c],  0.0f));
    hseg[idx] = f2bf(fmaxf(f0 * segW1[c] + f1 * segW1[128 + c] + segb1[c], 0.0f));
}

// feat = relu(temp + b_a + b_b) -> bf16
__global__ __launch_bounds__(256) void add2bias_relu(const float* __restrict__ temp,
                                                     const float* __restrict__ ba,
                                                     const float* __restrict__ bb,
                                                     ushort_t* __restrict__ outb,
                                                     int total) {
    long idx = (long)blockIdx.x * 256 + threadIdx.x;
    if (idx >= total) return;
    int c = (int)(idx & 127);
    outb[idx] = f2bf(fmaxf(temp[idx] + ba[c] + bb[c], 0.0f));
}

// h = relu(temp + b) -> bf16
__global__ __launch_bounds__(256) void bias_relu(const float* __restrict__ temp,
                                                 const float* __restrict__ b,
                                                 ushort_t* __restrict__ outb,
                                                 int total) {
    long idx = (long)blockIdx.x * 256 + threadIdx.x;
    if (idx >= total) return;
    int c = (int)(idx & 127);
    outb[idx] = f2bf(fmaxf(temp[idx] + b[c], 0.0f));
}

// feat = relu(temp + b + feat_old)  (in-place on bf16 feat buffer = residual)
__global__ __launch_bounds__(256) void residual_relu(const float* __restrict__ temp,
                                                     const float* __restrict__ b,
                                                     ushort_t* __restrict__ featb,
                                                     int total) {
    long idx = (long)blockIdx.x * 256 + threadIdx.x;
    if (idx >= total) return;
    int c = (int)(idx & 127);
    float v = temp[idx] + b[c] + bf2f(featb[idx]);
    featb[idx] = f2bf(fmaxf(v, 0.0f));
}

__global__ __launch_bounds__(256) void bf_to_f32(const ushort_t* __restrict__ in,
                                                 float* __restrict__ out,
                                                 int total) {
    long idx = (long)blockIdx.x * 256 + threadIdx.x;
    if (idx >= total) return;
    out[idx] = bf2f(in[idx]);
}

// ---------------------------------------------------------------------------
// Host orchestration
// ---------------------------------------------------------------------------
extern "C" void kernel_launch(void* const* d_in, const int* in_sizes, int n_in,
                              void* d_out, int out_size, void* d_ws, size_t ws_size,
                              hipStream_t stream) {
    (void)in_sizes; (void)n_in; (void)out_size; (void)ws_size;
    const int N = 200000, C = 128, S = 6, E = 200000, EL = 100000, NF = 4;

    const float* feats  = (const float*)d_in[0];
    const float* ctrs   = (const float*)d_in[1];
    const int* pre_u    = (const int*)d_in[2];
    const int* pre_v    = (const int*)d_in[3];
    const int* suc_u    = (const int*)d_in[4];
    const int* suc_v    = (const int*)d_in[5];
    const int* left_u   = (const int*)d_in[6];
    const int* left_v   = (const int*)d_in[7];
    const int* right_u  = (const int*)d_in[8];
    const int* right_v  = (const int*)d_in[9];
    const float* in_W1  = (const float*)d_in[10];
    const float* in_b1  = (const float*)d_in[11];
    const float* in_W2  = (const float*)d_in[12];
    const float* in_b2  = (const float*)d_in[13];
    const float* seg_W1 = (const float*)d_in[14];
    const float* seg_b1 = (const float*)d_in[15];
    const float* seg_W2 = (const float*)d_in[16];
    const float* seg_b2 = (const float*)d_in[17];
    const float* ctr_W  = (const float*)d_in[18];
    const float* pre_W  = (const float*)d_in[19];
    const float* suc_W  = (const float*)d_in[20];
    const float* left_W = (const float*)d_in[21];
    const float* right_W= (const float*)d_in[22];
    const float* norm_g = (const float*)d_in[23];
    const float* norm_b = (const float*)d_in[24];
    const float* c2_W1  = (const float*)d_in[25];
    const float* c2_b1  = (const float*)d_in[26];
    const float* c2_W2  = (const float*)d_in[27];
    const float* c2_b2  = (const float*)d_in[28];

    // workspace carving
    char* ws = (char*)d_ws;
    float*    temp  = (float*)ws;     ws += (size_t)N * C * sizeof(float);
    ushort_t* featb = (ushort_t*)ws;  ws += (size_t)N * C * sizeof(ushort_t);
    ushort_t* ha    = (ushort_t*)ws;  ws += (size_t)N * C * sizeof(ushort_t);
    ushort_t* hb    = (ushort_t*)ws;  ws += (size_t)N * C * sizeof(ushort_t);
    ushort_t* wswz  = (ushort_t*)ws;  // 70 matrices * 16384 bf16 = ~2.2 MB

    // --- weight pre-swizzle to bf16 WMMA-B layout (L2 resident afterwards) ---
    // slots: 0 in_W2 | 1 seg_W2 | 2..5 ctr | 6..29 pre | 30..53 suc
    //        54..57 left | 58..61 right | 62..65 c2_W1 | 66..69 c2_W2
    auto prep = [&](const float* src, int slot, int nmat) {
        long tot = (long)nmat * 16384;
        prep_weights<<<dim3((unsigned)((tot + 255) / 256)), dim3(256), 0, stream>>>(
            src, wswz + (size_t)slot * 16384, nmat);
    };
    prep(in_W2, 0, 1);   prep(seg_W2, 1, 1);
    prep(ctr_W, 2, 4);   prep(pre_W, 6, 24);   prep(suc_W, 30, 24);
    prep(left_W, 54, 4); prep(right_W, 58, 4);
    prep(c2_W1, 62, 4);  prep(c2_W2, 66, 4);

    const int total   = N * C;
    const dim3 pwGrid((unsigned)((total + 255) / 256)), pwBlk(256);
    const dim3 gGrid((unsigned)((N + 127) / 128)), gBlk(256);
    const dim3 eGrid((unsigned)((E + 127) / 128));
    const dim3 lGrid((unsigned)((EL + 127) / 128));
    const dim3 gnGrid((unsigned)((N + 7) / 8));

    // --- input stage: feat = relu(mlp(ctrs) + mlp(feats)) ---
    input_stage1<<<pwGrid, pwBlk, 0, stream>>>(feats, ctrs, in_W1, in_b1,
                                               seg_W1, seg_b1, ha, hb, total);
    wmma_edge_gemm<false><<<gGrid, gBlk, 0, stream>>>(ha, wswz + (size_t)0 * 16384,
                                                      nullptr, nullptr, temp, N);
    wmma_edge_gemm<true><<<gGrid, gBlk, 0, stream>>>(hb, wswz + (size_t)1 * 16384,
                                                     nullptr, nullptr, temp, N);
    add2bias_relu<<<pwGrid, pwBlk, 0, stream>>>(temp, in_b2, seg_b2, featb, total);

    // --- fusion rounds (featb doubles as the residual) ---
    for (int i = 0; i < NF; ++i) {
        // temp = feat @ ctr_W[i]
        wmma_edge_gemm<false><<<gGrid, gBlk, 0, stream>>>(
            featb, wswz + (size_t)(2 + i) * 16384, nullptr, nullptr, temp, N);
        // multi-scale pre / suc message passing: gather -> WMMA -> atomic scatter
        for (int s = 0; s < S; ++s) {
            wmma_edge_gemm<true><<<eGrid, gBlk, 0, stream>>>(
                featb, wswz + (size_t)(6 + i * 6 + s) * 16384,
                pre_u + (size_t)s * E, pre_v + (size_t)s * E, temp, E);
        }
        for (int s = 0; s < S; ++s) {
            wmma_edge_gemm<true><<<eGrid, gBlk, 0, stream>>>(
                featb, wswz + (size_t)(30 + i * 6 + s) * 16384,
                suc_u + (size_t)s * E, suc_v + (size_t)s * E, temp, E);
        }
        wmma_edge_gemm<true><<<lGrid, gBlk, 0, stream>>>(
            featb, wswz + (size_t)(54 + i) * 16384, left_u, left_v, temp, EL);
        wmma_edge_gemm<true><<<lGrid, gBlk, 0, stream>>>(
            featb, wswz + (size_t)(58 + i) * 16384, right_u, right_v, temp, EL);
        // feat' = relu(GroupNorm(temp))
        gn_relu<<<gnGrid, gBlk, 0, stream>>>(temp, norm_g + (size_t)i * C,
                                             norm_b + (size_t)i * C, ha, N);
        // 2-layer MLP
        wmma_edge_gemm<false><<<gGrid, gBlk, 0, stream>>>(
            ha, wswz + (size_t)(62 + i) * 16384, nullptr, nullptr, temp, N);
        bias_relu<<<pwGrid, pwBlk, 0, stream>>>(temp, c2_b1 + (size_t)i * C, hb, total);
        wmma_edge_gemm<false><<<gGrid, gBlk, 0, stream>>>(
            hb, wswz + (size_t)(66 + i) * 16384, nullptr, nullptr, temp, N);
        // feat = relu(mlp_out + b2 + residual)
        residual_relu<<<pwGrid, pwBlk, 0, stream>>>(temp, c2_b2 + (size_t)i * C,
                                                    featb, total);
    }

    // final fp32 output
    bf_to_f32<<<pwGrid, pwBlk, 0, stream>>>(featb, (float*)d_out, total);
}
